// MultiHeadSelfAttention_43516608643698
// MI455X (gfx1250) — matrix operand order repaired and checked by reading, not compiled, not run on hardware
//
#include <hip/hip_runtime.h>
#include <hip/hip_bf16.h>

typedef __bf16 bf16_t;
typedef __attribute__((ext_vector_type(16))) __bf16 v16bf;
typedef __attribute__((ext_vector_type(8)))  __bf16 v8bf;
typedef __attribute__((ext_vector_type(8)))  float  v8f;
typedef __attribute__((ext_vector_type(4)))  int    v4i;

#define BDIM 2
#define SDIM 2048
#define EDIM 2048
#define HDIM 16
#define DDIM 128

#if defined(__has_builtin)
#if __has_builtin(__builtin_amdgcn_global_load_async_to_lds_b128)
#define HAVE_ASYNC_LDS 1
#endif
#if __has_builtin(__builtin_amdgcn_s_wait_asynccnt)
#define HAVE_WAIT_ASYNC 1
#endif
#endif

__device__ __forceinline__ v8f wmma_bf16(v16bf a, v16bf b, v8f c) {
  // D = A(16x32) * B(32x16) + C, fp32 accumulate
  return __builtin_amdgcn_wmma_f32_16x16x32_bf16(false, a, false, b, (short)0, c, false, false);
}

// ---- 16-byte global->LDS copy: async on CDNA5 (ASYNCcnt), reg-copy fallback ----
__device__ __forceinline__ void copy16_g2l(const bf16_t* g, bf16_t* l) {
#ifdef HAVE_ASYNC_LDS
  __builtin_amdgcn_global_load_async_to_lds_b128(
      (__attribute__((address_space(1))) v4i*)g,
      (__attribute__((address_space(3))) v4i*)l, 0, 0);
#else
  *(v8bf*)l = *(const v8bf*)g;
#endif
}

__device__ __forceinline__ void wait_async0() {
#ifdef HAVE_ASYNC_LDS
#ifdef HAVE_WAIT_ASYNC
  __builtin_amdgcn_s_wait_asynccnt(0);
#else
  asm volatile("s_wait_asynccnt 0x0" ::: "memory");
#endif
#endif
}

// A-operand tile from global: rows row0..row0+15, K-chunk [k0,k0+32).
// Contiguous-k A layout used here: lane L (r=L%16, half=L/16), element i holds
// K = k0 + 16*half + i, M = r.
__device__ __forceinline__ v16bf load_tileA(const bf16_t* __restrict__ base, int ld,
                                            int row0, int k0) {
  int lane = threadIdx.x & 31;
  int r = lane & 15, half = lane >> 4;
  const bf16_t* p = base + (size_t)(row0 + r) * (size_t)ld + (size_t)(k0 + 16 * half);
  v8bf lo = *(const v8bf*)p;
  v8bf hi = *(const v8bf*)(p + 8);
  v16bf out;
#pragma unroll
  for (int i = 0; i < 8; ++i) { out[i] = lo[i]; out[i + 8] = hi[i]; }
  return out;
}

// B-operand tile from global: B[K,N] = Src[row0+N, k].
// Contiguous-k B layout used here: lane L (col=L%16, half=L/16), element i holds K = k0 + 16*half + i.
__device__ __forceinline__ v16bf load_tileB(const bf16_t* __restrict__ base, int ld,
                                            int row0, int k0) {
  int lane = threadIdx.x & 31;
  int col = lane & 15, half = lane >> 4;
  const bf16_t* p = base + (size_t)(row0 + col) * (size_t)ld + (size_t)(k0 + 16 * half);
  return *(const v16bf*)p;
}

// ---- LDS tile readers (row stride padded to 40 elements = 80B to spread banks) ----
#define LDS_STRIDE 40

__device__ __forceinline__ v16bf lds_tileA(const bf16_t* base /* tile row0 applied */) {
  int lane = threadIdx.x & 31;
  int r = lane & 15, half = lane >> 4;
  const bf16_t* p = base + r * LDS_STRIDE + 8 * half;
  v8bf lo = *(const v8bf*)p;
  v8bf hi = *(const v8bf*)(p + 16);
  v16bf out;
#pragma unroll
  for (int i = 0; i < 8; ++i) { out[i] = lo[i]; out[i + 8] = hi[i]; }
  return out;
}

__device__ __forceinline__ v16bf lds_tileB(const bf16_t* base) {
  int lane = threadIdx.x & 31;
  int col = lane & 15, half = lane >> 4;
  const bf16_t* p = base + col * LDS_STRIDE + 8 * half;
  v8bf lo = *(const v8bf*)p;        // 16B-aligned halves -> ds_load_b128 pair
  v8bf hi = *(const v8bf*)(p + 16);
  v16bf out;
#pragma unroll
  for (int i = 0; i < 8; ++i) { out[i] = lo[i]; out[i + 8] = hi[i]; }
  return out;
}

__global__ void cvt_f32_to_bf16(const float* __restrict__ src, bf16_t* __restrict__ dst,
                                long long n) {
  long long i = (long long)blockIdx.x * blockDim.x + threadIdx.x;
  if (i < n) dst[i] = (bf16_t)src[i];
}

// C[m,n] = sum_k A_bf16[m,k] * W_bf16[n,k]  (fp32 out). M%128==0, N%128==0, K%32==0.
// 256 threads = 8 waves (2x4 wave grid). Block tile 128x128, double-buffered LDS
// staging fed by async global->LDS copies. Each wave: 64(M) x 32(N) = 8 WMMA/K-step.
__global__ __launch_bounds__(256) void gemm_bf16_nt(const bf16_t* __restrict__ A,
                                                    const bf16_t* __restrict__ W,
                                                    float* __restrict__ C,
                                                    int M, int N, int K) {
  __shared__ bf16_t lA[2][128 * LDS_STRIDE];
  __shared__ bf16_t lB[2][128 * LDS_STRIDE];

  int nb = N >> 7;
  int bm = blockIdx.x / nb;
  int bn = blockIdx.x - bm * nb;
  int m_base = bm << 7, n_base = bn << 7;
  int tid  = threadIdx.x;
  int wave = tid >> 5, lane = tid & 31;
  int wm = wave >> 2, wn = wave & 3;            // 2(M) x 4(N)
  int col = lane & 15, half = lane >> 4;

  // stage one 128x32 K-panel of A and of W into LDS buffer `buf`
  auto stage = [&](int buf, int k0) {
#pragma unroll
    for (int c = tid; c < 512; c += 256) {      // 128 rows x 4 chunks of 8 elems
      int row = c >> 2, ch = (c & 3) << 3;
      copy16_g2l(A + (size_t)(m_base + row) * (size_t)K + (size_t)(k0 + ch),
                 &lA[buf][row * LDS_STRIDE + ch]);
    }
#pragma unroll
    for (int c = tid; c < 512; c += 256) {
      int row = c >> 2, ch = (c & 3) << 3;
      copy16_g2l(W + (size_t)(n_base + row) * (size_t)K + (size_t)(k0 + ch),
                 &lB[buf][row * LDS_STRIDE + ch]);
    }
  };

  v8f acc[4][2];
#pragma unroll
  for (int i = 0; i < 4; ++i)
#pragma unroll
    for (int j = 0; j < 2; ++j)
#pragma unroll
      for (int r = 0; r < 8; ++r) acc[i][j][r] = 0.f;

  stage(0, 0);
  int nk = K >> 5;
  for (int ki = 0; ki < nk; ++ki) {
    int buf = ki & 1;
    wait_async0();          // my async copies into `buf` are done
    __syncthreads();        // everyone's are; prior reads of buf^1 also retired
    if (ki + 1 < nk) stage(buf ^ 1, (ki + 1) << 5);

    const bf16_t* bufA = lA[buf];
    const bf16_t* bufB = lB[buf];
    v16bf at[4], bt[2];
#pragma unroll
    for (int i = 0; i < 4; ++i)
      at[i] = lds_tileA(bufA + ((wm << 6) + (i << 4)) * LDS_STRIDE);
#pragma unroll
    for (int j = 0; j < 2; ++j)
      bt[j] = lds_tileB(bufB + ((wn << 5) + (j << 4)) * LDS_STRIDE);
#pragma unroll
    for (int i = 0; i < 4; ++i)
#pragma unroll
      for (int j = 0; j < 2; ++j)
        acc[i][j] = wmma_bf16(at[i], bt[j], acc[i][j]);
  }

  // C layout: acc[i][j][r] at (M = m_base + wm*64 + 16*i + r + 8*half,
  //                            N = n_base + wn*32 + 16*j + col)
#pragma unroll
  for (int i = 0; i < 4; ++i)
#pragma unroll
    for (int j = 0; j < 2; ++j)
#pragma unroll
      for (int r = 0; r < 8; ++r) {
        size_t mg = (size_t)(m_base + (wm << 6) + (i << 4) + r + 8 * half);
        size_t ng = (size_t)(n_base + (wn << 5) + (j << 4) + col);
        C[mg * (size_t)N + ng] = acc[i][j][r];
      }
}

// RoPE q,k; emit bf16 q_rot/k_rot (B,H,S,D); raw fp32 k,v outputs (B,H,S,D);
// bf16 V^T (B,H,D,S). One thread per (b,h,s,pair).
__global__ void rope_pack(const float* __restrict__ yq, const float* __restrict__ yk,
                          const float* __restrict__ yv,
                          bf16_t* __restrict__ q_rot, bf16_t* __restrict__ k_rot,
                          bf16_t* __restrict__ vt,
                          float* __restrict__ k_out, float* __restrict__ v_out) {
  long long tid = (long long)blockIdx.x * blockDim.x + threadIdx.x;
  const long long total = (long long)BDIM * HDIM * SDIM * (DDIM / 2);
  if (tid >= total) return;
  int p = (int)(tid & (DDIM / 2 - 1));
  long long t = tid >> 6;                 // D/2 = 64
  int s = (int)(t & (SDIM - 1)); t >>= 11;
  int h = (int)(t & (HDIM - 1)); t >>= 4;
  int b = (int)t;
  int d0 = p << 1, d1 = d0 + 1;
  size_t src = ((size_t)b * SDIM + (size_t)s) * EDIM + (size_t)h * DDIM;
  size_t dst = (((size_t)b * HDIM + h) * SDIM + (size_t)s) * DDIM;
  float inv = __powf(10000.f, -((float)d0) / (float)DDIM);
  float ang = (float)s * inv;
  float cs = __cosf(ang), sn = __sinf(ang);
  float q1 = yq[src + d0], q2 = yq[src + d1];
  q_rot[dst + d0] = (bf16_t)(q1 * cs - q2 * sn);
  q_rot[dst + d1] = (bf16_t)(q1 * sn + q2 * cs);
  float k1 = yk[src + d0], k2 = yk[src + d1];
  k_rot[dst + d0] = (bf16_t)(k1 * cs - k2 * sn);
  k_rot[dst + d1] = (bf16_t)(k1 * sn + k2 * cs);
  k_out[dst + d0] = k1; k_out[dst + d1] = k2;
  float v1 = yv[src + d0], v2 = yv[src + d1];
  v_out[dst + d0] = v1; v_out[dst + d1] = v2;
  size_t vbase = (((size_t)b * HDIM + h) * DDIM) * SDIM + (size_t)s;
  vt[vbase + (size_t)d0 * SDIM] = (bf16_t)v1;
  vt[vbase + (size_t)d1 * SDIM] = (bf16_t)v2;
}

// Flash attention, one wave per 16-query tile. Computes S^T = K_rot * Q_rot^T so
// softmax stats are per lane-column (reduce across halves with shfl_xor 16), and
// accumulates O^T = V^T * P. Output written as bf16 (B,S,E) for the Wo GEMM.
__global__ __launch_bounds__(128) void attn_fwd(const bf16_t* __restrict__ q_rot,
                                                const bf16_t* __restrict__ k_rot,
                                                const bf16_t* __restrict__ vt,
                                                bf16_t* __restrict__ ob) {
  const float scale = 0.08838834764831843f;   // 1/sqrt(128)
  const float NEG = -3.0e38f;
  int wave = threadIdx.x >> 5;
  int lane = threadIdx.x & 31;
  int col = lane & 15, half = lane >> 4;
  int nqb = SDIM >> 6;
  int bh = blockIdx.x / nqb;
  int q0 = ((blockIdx.x - bh * nqb) << 6) + (wave << 4);
  const bf16_t* Q  = q_rot + (size_t)bh * SDIM * DDIM;
  const bf16_t* Kp = k_rot + (size_t)bh * SDIM * DDIM;
  const bf16_t* Vt = vt    + (size_t)bh * SDIM * DDIM;

  v16bf qv[4];
#pragma unroll
  for (int c = 0; c < 4; ++c) qv[c] = load_tileB(Q, DDIM, q0, c * 32);

  v8f acc[8];                                  // O^T: 8 tiles over D (M=d, N=q)
#pragma unroll
  for (int j = 0; j < 8; ++j)
#pragma unroll
    for (int r = 0; r < 8; ++r) acc[j][r] = 0.f;

  float m_i = NEG, l_i = 0.f;
  int nkv = q0 + 16;
  for (int kv0 = 0; kv0 < nkv; kv0 += 32) {
    v8f s0, s1;
#pragma unroll
    for (int r = 0; r < 8; ++r) { s0[r] = 0.f; s1[r] = 0.f; }
#pragma unroll
    for (int c = 0; c < 4; ++c) {
      v16bf ka = load_tileA(Kp, DDIM, kv0, c * 32);
      s0 = wmma_bf16(ka, qv[c], s0);
    }
#pragma unroll
    for (int c = 0; c < 4; ++c) {
      v16bf kb = load_tileA(Kp, DDIM, kv0 + 16, c * 32);
      s1 = wmma_bf16(kb, qv[c], s1);
    }
    // scale + causal mask. S^T tile: c[r] holds (kv = kv0 + 16*tile + r + 8*half, q = q0+col)
    int qg = q0 + col;
    float t0[8], t1[8];
#pragma unroll
    for (int r = 0; r < 8; ++r) {
      int kg0 = kv0 + r + 8 * half;
      int kg1 = kg0 + 16;
      t0[r] = (kg0 <= qg) ? s0[r] * scale : NEG;
      t1[r] = (kg1 <= qg) ? s1[r] * scale : NEG;
    }
    float mloc = t0[0];
#pragma unroll
    for (int r = 0; r < 8; ++r) { mloc = fmaxf(mloc, t0[r]); mloc = fmaxf(mloc, t1[r]); }
    mloc = fmaxf(mloc, __shfl_xor(mloc, 16, 32));
    float m_new = fmaxf(m_i, mloc);
    float alpha = __expf(m_i - m_new);
    float p0[8], p1[8], psum = 0.f;
#pragma unroll
    for (int r = 0; r < 8; ++r) {
      p0[r] = __expf(t0[r] - m_new);
      p1[r] = __expf(t1[r] - m_new);
      psum += p0[r] + p1[r];
    }
    psum += __shfl_xor(psum, 16, 32);
    l_i = l_i * alpha + psum;
    m_i = m_new;
    // Repack P into B-operand layout: lane(half) element i holds kv = kv0 + 16*half + i.
    v16bf pb;
#pragma unroll
    for (int r = 0; r < 8; ++r) {
      float o0 = __shfl_xor(p0[r], 16, 32);
      float o1 = __shfl_xor(p1[r], 16, 32);
      float e_lo = half ? o1 : p0[r];
      float e_hi = half ? p1[r] : o0;
      pb[r]     = (bf16_t)e_lo;
      pb[r + 8] = (bf16_t)e_hi;
    }
    // rescale and accumulate O^T += V^T * P
#pragma unroll
    for (int j = 0; j < 8; ++j) {
#pragma unroll
      for (int r = 0; r < 8; ++r) acc[j][r] *= alpha;
      v16bf va = load_tileA(Vt, SDIM, j << 4, kv0);
      acc[j] = wmma_bf16(va, pb, acc[j]);
    }
  }
  // normalize and store: ob[b, q, h*D + d], d = 16*j + r + 8*half, q = q0+col
  float linv = 1.0f / l_i;
  int b = bh / HDIM, h = bh - b * HDIM;
  size_t base = ((size_t)(b * SDIM + (q0 + col))) * (size_t)EDIM + (size_t)h * DDIM;
#pragma unroll
  for (int j = 0; j < 8; ++j)
#pragma unroll
    for (int r = 0; r < 8; ++r)
      ob[base + (j << 4) + r + 8 * half] = (bf16_t)(acc[j][r] * linv);
}

extern "C" void kernel_launch(void* const* d_in, const int* in_sizes, int n_in,
                              void* d_out, int out_size, void* d_ws, size_t ws_size,
                              hipStream_t stream) {
  (void)in_sizes; (void)n_in; (void)out_size; (void)ws_size;
  const float* x  = (const float*)d_in[0];
  const float* Wq = (const float*)d_in[1];
  const float* Wk = (const float*)d_in[2];
  const float* Wv = (const float*)d_in[3];
  const float* Wo = (const float*)d_in[4];

  const size_t NBSE = (size_t)BDIM * SDIM * EDIM;     // 8M elements
  float* out      = (float*)d_out;
  float* attn_out = out;
  float* k_out    = out + NBSE;
  float* v_out    = out + 2 * NBSE;

  char* ws = (char*)d_ws;
  const size_t MB = 1ull << 20;
  bf16_t* xb   = (bf16_t*)(ws + 0);        // 16 MB
  bf16_t* wqb  = (bf16_t*)(ws + 16 * MB);  // 8 MB each
  bf16_t* wkb  = (bf16_t*)(ws + 24 * MB);
  bf16_t* wvb  = (bf16_t*)(ws + 32 * MB);
  bf16_t* wob  = (bf16_t*)(ws + 40 * MB);
  float*  yq   = (float*)(ws + 48 * MB);   // 32 MB each
  float*  yk   = (float*)(ws + 80 * MB);
  float*  yv   = (float*)(ws + 112 * MB);
  bf16_t* qrot = (bf16_t*)(ws + 144 * MB); // 16 MB each
  bf16_t* krot = (bf16_t*)(ws + 160 * MB);
  bf16_t* vtb  = (bf16_t*)(ws + 176 * MB);
  bf16_t* ob   = (bf16_t*)(ws + 0);        // alias xb (dead after projections)

  const long long nX = (long long)NBSE, nW = (long long)EDIM * EDIM;
  cvt_f32_to_bf16<<<(unsigned)((nX + 255) / 256), 256, 0, stream>>>(x,  xb,  nX);
  cvt_f32_to_bf16<<<(unsigned)((nW + 255) / 256), 256, 0, stream>>>(Wq, wqb, nW);
  cvt_f32_to_bf16<<<(unsigned)((nW + 255) / 256), 256, 0, stream>>>(Wk, wkb, nW);
  cvt_f32_to_bf16<<<(unsigned)((nW + 255) / 256), 256, 0, stream>>>(Wv, wvb, nW);
  cvt_f32_to_bf16<<<(unsigned)((nW + 255) / 256), 256, 0, stream>>>(Wo, wob, nW);

  const int M = BDIM * SDIM, N = EDIM, K = EDIM;
  dim3 gblocks((M / 128) * (N / 128));
  gemm_bf16_nt<<<gblocks, 256, 0, stream>>>(xb, wqb, yq, M, N, K);
  gemm_bf16_nt<<<gblocks, 256, 0, stream>>>(xb, wkb, yk, M, N, K);
  gemm_bf16_nt<<<gblocks, 256, 0, stream>>>(xb, wvb, yv, M, N, K);

  long long nrp = (long long)BDIM * HDIM * SDIM * (DDIM / 2);
  rope_pack<<<(unsigned)((nrp + 255) / 256), 256, 0, stream>>>(yq, yk, yv, qrot, krot, vtb,
                                                               k_out, v_out);

  int ablocks = BDIM * HDIM * (SDIM / 64);
  attn_fwd<<<ablocks, 128, 0, stream>>>(qrot, krot, vtb, ob);

  gemm_bf16_nt<<<gblocks, 256, 0, stream>>>(ob, wob, attn_out, M, N, K);
}
